// AttnFFN_17085379903863
// MI455X (gfx1250) — matrix operand order repaired and checked by reading, not compiled, not run on hardware
//
#include <hip/hip_runtime.h>

typedef __bf16 bf16;
typedef __attribute__((ext_vector_type(16))) __bf16 v16bf;
typedef __attribute__((ext_vector_type(8)))  __bf16 v8bf;
typedef __attribute__((ext_vector_type(8)))  float  v8f;

#define NTOK 256
#define HEADS 8
#define KD 32
#define DHEAD 128
#define DIM 384
#define DH 1024
#define HID 1536

// ---- WMMA fragment loaders (CDNA5 ISA 7.12.2, 16-bit operands) ----
// A (16x32 MxK): lane owns K = kbase..kbase+15, kbase = (lane&16)?16:0
__device__ __forceinline__ v16bf load_afrag(const bf16* base, int lane) {
  return *(const v16bf*)(base + ((lane & 16) ? 16 : 0));
}
// B (32x16 KxN): lane owns K = kbase..kbase+15 at one column, kbase = (lane&16)?16:0
__device__ __forceinline__ v16bf load_bfrag(const bf16* base, int lane) {
  return *(const v16bf*)(base + ((lane & 16) ? 16 : 0));
}

// ------------------------------------------------------------------
__global__ void cvt_bf16(const float* __restrict__ in, bf16* __restrict__ out, int n) {
  int i = blockIdx.x * blockDim.x + threadIdx.x;
  if (i < n) out[i] = (bf16)in[i];
}

// channel-major f32 [b][C][256] -> token-major bf16 [b][256][C]
__global__ void cvt_tr_bf16(const float* __restrict__ in, bf16* __restrict__ out,
                            int C, int total) {
  int i = blockIdx.x * blockDim.x + threadIdx.x;
  if (i >= total) return;
  int c = i % C;
  int n = (i / C) & 255;
  int b = i / (C * 256);
  out[i] = (bf16)in[((size_t)b * C + c) * 256 + n];
}

__global__ void bn_fold(const float* __restrict__ bias, const float* __restrict__ g,
                        const float* __restrict__ beta, const float* __restrict__ m,
                        const float* __restrict__ v, float* __restrict__ scale,
                        float* __restrict__ shift, int C) {
  int i = blockIdx.x * blockDim.x + threadIdx.x;
  if (i >= C) return;
  float s = g[i] * rsqrtf(v[i] + 1e-5f);
  scale[i] = s;
  shift[i] = (bias[i] - m[i]) * s + beta[i];
}

// Depthwise 3x3 SAME, channel-major layout: out = acc + p0[c]
__global__ void dwconv3_c(const bf16* __restrict__ in, const float* __restrict__ w9,
                          const float* __restrict__ p0, bf16* __restrict__ out,
                          int C, int total) {
  int i = blockIdx.x * blockDim.x + threadIdx.x;
  if (i >= total) return;
  int n = i & 255;
  int c = (i >> 8) % C;
  int y = n >> 4, x = n & 15;
  const bf16* plane = in + (size_t)(i - n);
  float acc = 0.f;
  #pragma unroll
  for (int dy = -1; dy <= 1; ++dy)
    #pragma unroll
    for (int dx = -1; dx <= 1; ++dx) {
      int yy = y + dy, xx = x + dx;
      if (yy >= 0 && yy < 16 && xx >= 0 && xx < 16)
        acc += w9[c * 9 + (dy + 1) * 3 + (dx + 1)] * (float)plane[yy * 16 + xx];
    }
  out[i] = (bf16)(acc + p0[c]);
}

// Depthwise 3x3 SAME, token-major layout: out = relu(acc*p0[c] + p1[c])
__global__ void dwconv3_t(const bf16* __restrict__ in, const float* __restrict__ w9,
                          const float* __restrict__ p0, const float* __restrict__ p1,
                          bf16* __restrict__ out, int C, int total) {
  int i = blockIdx.x * blockDim.x + threadIdx.x;
  if (i >= total) return;
  int c = i % C;
  int n = (i / C) & 255;
  int b = i / (C * 256);
  int y = n >> 4, x = n & 15;
  const bf16* base = in + (size_t)b * 256 * C;
  float acc = 0.f;
  #pragma unroll
  for (int dy = -1; dy <= 1; ++dy)
    #pragma unroll
    for (int dx = -1; dx <= 1; ++dx) {
      int yy = y + dy, xx = x + dx;
      if (yy >= 0 && yy < 16 && xx >= 0 && xx < 16)
        acc += w9[c * 9 + (dy + 1) * 3 + (dx + 1)] * (float)base[(yy * 16 + xx) * C + c];
    }
  out[i] = (bf16)fmaxf(acc * p0[c] + p1[c], 0.f);
}

// ------------------------------------------------------------------
// Batched GEMM: Y[b](M x 256) = W(M x K) @ Xt[b](256 x K)^T.
// Ping-pong pipelined K loop (P/Q fragment sets refilled in place -> no reg copies).
// epi 0:+bias  1:+bias+resid  2:relu(acc*scale+shift)  3:acc*scale+shift+resid
__launch_bounds__(256)
__global__ void gemm_bf16(const bf16* __restrict__ W, const bf16* __restrict__ Xt,
                          const float* __restrict__ p0, const float* __restrict__ p1,
                          const float* __restrict__ resid,
                          float* __restrict__ outF, bf16* __restrict__ outB,
                          int M, int K, int epi, int obChan, int ofChan, int rChan) {
  const int lane = threadIdx.x & 31;
  const int wave = threadIdx.x >> 5;
  const int b = blockIdx.z;
  const int m0 = blockIdx.y * 128 + wave * 16;
  const int n0 = blockIdx.x * 64;
  const int col = lane & 15;

  const bf16* wrow = W + (size_t)(m0 + col) * K;
  const bf16* xt0 = Xt + ((size_t)b * NTOK + n0 + col) * K;

  v8f acc[4] = {};
  // prologue: P holds k=0, Q holds k=32
  v16bf aP = load_afrag(wrow, lane);
  v16bf bP[4];
  #pragma unroll
  for (int nt = 0; nt < 4; ++nt) bP[nt] = load_bfrag(xt0 + (size_t)nt * 16 * K, lane);
  v16bf aQ = aP;
  v16bf bQ[4];
  if (K > 32) {
    aQ = load_afrag(wrow + 32, lane);
    #pragma unroll
    for (int nt = 0; nt < 4; ++nt) bQ[nt] = load_bfrag(xt0 + (size_t)nt * 16 * K + 32, lane);
  } else {
    #pragma unroll
    for (int nt = 0; nt < 4; ++nt) bQ[nt] = bP[nt];
  }

  int k0 = 0;
  #pragma unroll 1
  for (; k0 + 64 < K; k0 += 64) {
    // consume P (k0), refill P in place from k0+64
    #pragma unroll
    for (int nt = 0; nt < 4; ++nt)
      acc[nt] = __builtin_amdgcn_wmma_f32_16x16x32_bf16(
          false, aP, false, bP[nt], (short)0, acc[nt], false, false);
    aP = load_afrag(wrow + k0 + 64, lane);
    #pragma unroll
    for (int nt = 0; nt < 4; ++nt)
      bP[nt] = load_bfrag(xt0 + (size_t)nt * 16 * K + k0 + 64, lane);
    // consume Q (k0+32), refill Q in place from k0+96
    #pragma unroll
    for (int nt = 0; nt < 4; ++nt)
      acc[nt] = __builtin_amdgcn_wmma_f32_16x16x32_bf16(
          false, aQ, false, bQ[nt], (short)0, acc[nt], false, false);
    if (k0 + 96 < K) {
      aQ = load_afrag(wrow + k0 + 96, lane);
      #pragma unroll
      for (int nt = 0; nt < 4; ++nt)
        bQ[nt] = load_bfrag(xt0 + (size_t)nt * 16 * K + k0 + 96, lane);
      __builtin_prefetch(wrow + k0 + 128, 0, 1);
      __builtin_prefetch(xt0 + k0 + 128, 0, 1);
    }
  }
  // tail: P = k0, optionally Q = k0+32
  #pragma unroll
  for (int nt = 0; nt < 4; ++nt)
    acc[nt] = __builtin_amdgcn_wmma_f32_16x16x32_bf16(
        false, aP, false, bP[nt], (short)0, acc[nt], false, false);
  if (k0 + 32 < K) {
    #pragma unroll
    for (int nt = 0; nt < 4; ++nt)
      acc[nt] = __builtin_amdgcn_wmma_f32_16x16x32_bf16(
          false, aQ, false, bQ[nt], (short)0, acc[nt], false, false);
  }

  const int rhalf = (lane >> 4) << 3;
  #pragma unroll
  for (int nt = 0; nt < 4; ++nt) {
    const int nc = n0 + nt * 16 + col;
    float vals[8];
    #pragma unroll
    for (int r = 0; r < 8; ++r) {
      const int ch = m0 + r + rhalf;
      float v = acc[nt][r];
      if (epi == 0) v += p0[ch];
      else if (epi == 1) {
        size_t ridx = rChan ? ((size_t)b * M + ch) * NTOK + nc
                            : ((size_t)b * NTOK + nc) * M + ch;
        v += p0[ch] + resid[ridx];
      } else if (epi == 2) {
        v = fmaxf(v * p0[ch] + p1[ch], 0.f);
      } else {
        size_t ridx = rChan ? ((size_t)b * M + ch) * NTOK + nc
                            : ((size_t)b * NTOK + nc) * M + ch;
        v = v * p0[ch] + p1[ch] + resid[ridx];
      }
      vals[r] = v;
    }
    if (outF) {
      if (ofChan) {
        #pragma unroll
        for (int r = 0; r < 8; ++r)
          outF[((size_t)b * M + m0 + r + rhalf) * NTOK + nc] = vals[r];
      } else {
        v8f pk;
        #pragma unroll
        for (int r = 0; r < 8; ++r) pk[r] = vals[r];
        *(v8f*)(outF + ((size_t)b * NTOK + nc) * M + m0 + rhalf) = pk;
      }
    }
    if (outB) {
      if (obChan) {
        #pragma unroll
        for (int r = 0; r < 8; ++r)
          outB[((size_t)b * M + m0 + r + rhalf) * NTOK + nc] = (bf16)vals[r];
      } else {
        v8bf pk;
        #pragma unroll
        for (int r = 0; r < 8; ++r) pk[r] = (bf16)vals[r];
        *(v8bf*)(outB + ((size_t)b * NTOK + nc) * M + m0 + rhalf) = pk;
      }
    }
  }
}

// ------------------------------------------------------------------
// Cross-head 8x8 mix over LDS scores, register-staged (2 rows / iter)
__device__ __forceinline__ void head_mix(bf16* S, const float* Tw, const float* Tb) {
  const int m = threadIdx.x;
  #pragma unroll 1
  for (int it = 0; it < 8; ++it) {
    int na = it * 2, nb = it * 2 + 1;
    float in0[8], in1[8], o0[8], o1[8];
    #pragma unroll
    for (int g = 0; g < 8; ++g) {
      in0[g] = (float)S[(g * 16 + na) * NTOK + m];
      in1[g] = (float)S[(g * 16 + nb) * NTOK + m];
    }
    #pragma unroll
    for (int o = 0; o < 8; ++o) {
      float s0 = Tb[o], s1 = Tb[o];
      #pragma unroll
      for (int g = 0; g < 8; ++g) {
        s0 += Tw[o * 8 + g] * in0[g];
        s1 += Tw[o * 8 + g] * in1[g];
      }
      o0[o] = s0; o1[o] = s1;
    }
    __syncthreads();
    #pragma unroll
    for (int o = 0; o < 8; ++o) {
      S[(o * 16 + na) * NTOK + m] = (bf16)o0[o];
      S[(o * 16 + nb) * NTOK + m] = (bf16)o1[o];
    }
    __syncthreads();
  }
}

// Fused attention. Q,K token-major [b][n][256]; V channel-major [b][1024][m];
// U token-major [b][n][1024]. grid = B*16, 256 thr, wave == head.
__launch_bounds__(256)
__global__ void attn_fused(const bf16* __restrict__ Qt, const bf16* __restrict__ Kt,
                           const bf16* __restrict__ V, const float* __restrict__ ab,
                           const int* __restrict__ bidx,
                           const float* __restrict__ th1w, const float* __restrict__ th1b,
                           const float* __restrict__ th2w, const float* __restrict__ th2b,
                           const bf16* __restrict__ vloc, bf16* __restrict__ U) {
  __shared__ bf16 S[HEADS * 16 * NTOK];  // 64 KB of the WGP's 320 KB LDS
  __shared__ float sAb[HEADS * NTOK];
  __shared__ float sT1[64], sT2[64], sB1[8], sB2[8];
  const int lane = threadIdx.x & 31, wave = threadIdx.x >> 5;
  const int b = blockIdx.x >> 4;
  const int n0 = (blockIdx.x & 15) * 16;
  const int h = wave, col = lane & 15, rhalf = (lane >> 4) << 3;

  for (int i = threadIdx.x; i < HEADS * NTOK; i += 256) sAb[i] = ab[i];
  if (threadIdx.x < 64) { sT1[threadIdx.x] = th1w[threadIdx.x]; sT2[threadIdx.x] = th2w[threadIdx.x]; }
  if (threadIdx.x < 8)  { sB1[threadIdx.x] = th1b[threadIdx.x]; sB2[threadIdx.x] = th2b[threadIdx.x]; }
  __syncthreads();

  // ---- S_h = scale * Q_h^T K_h + bias : ping-pong pipelined K-tile loads ----
  const float scale = 0.17677669529663687f;  // kd^-0.5
  {
    const bf16* kbase = Kt + (size_t)b * NTOK * 256 + h * KD;
    v16bf a = load_afrag(Qt + ((size_t)b * NTOK + n0 + col) * 256 + h * KD, lane);
    v16bf bP = load_bfrag(kbase + (size_t)(0 + col) * 256, lane);
    v16bf bQ = load_bfrag(kbase + (size_t)(16 + col) * 256, lane);
    #pragma unroll 1
    for (int mt = 0; mt < 16; mt += 2) {
      // tile mt (set P)
      v8f c = {};
      c = __builtin_amdgcn_wmma_f32_16x16x32_bf16(false, a, false, bP, (short)0, c, false, false);
      if (mt + 2 < 16) bP = load_bfrag(kbase + (size_t)((mt + 2) * 16 + col) * 256, lane);
      #pragma unroll
      for (int r = 0; r < 8; ++r) {
        int nl = r + rhalf, m = mt * 16 + col;
        float bias = sAb[h * NTOK + bidx[(size_t)(n0 + nl) * NTOK + m]];
        S[(h * 16 + nl) * NTOK + m] = (bf16)(c[r] * scale + bias);
      }
      // tile mt+1 (set Q)
      v8f d = {};
      d = __builtin_amdgcn_wmma_f32_16x16x32_bf16(false, a, false, bQ, (short)0, d, false, false);
      if (mt + 3 < 16) bQ = load_bfrag(kbase + (size_t)((mt + 3) * 16 + col) * 256, lane);
      #pragma unroll
      for (int r = 0; r < 8; ++r) {
        int nl = r + rhalf, m = (mt + 1) * 16 + col;
        float bias = sAb[h * NTOK + bidx[(size_t)(n0 + nl) * NTOK + m]];
        S[(h * 16 + nl) * NTOK + m] = (bf16)(d[r] * scale + bias);
      }
    }
  }
  __syncthreads();

  head_mix(S, sT1, sB1);

  // ---- softmax: wave owns rows [wave*16, wave*16+16); lane owns 8 contiguous cols ----
  for (int r = wave * 16; r < wave * 16 + 16; ++r) {
    bf16* row = &S[r * NTOK];
    v8bf chunk = *(v8bf*)(row + lane * 8);
    float vals[8];
    float mx = -3.4e38f;
    #pragma unroll
    for (int e = 0; e < 8; ++e) { vals[e] = (float)chunk[e]; mx = fmaxf(mx, vals[e]); }
    #pragma unroll
    for (int off = 16; off; off >>= 1) mx = fmaxf(mx, __shfl_xor(mx, off, 32));
    float sum = 0.f;
    #pragma unroll
    for (int e = 0; e < 8; ++e) { vals[e] = __expf(vals[e] - mx); sum += vals[e]; }
    #pragma unroll
    for (int off = 16; off; off >>= 1) sum += __shfl_xor(sum, off, 32);
    float inv = 1.f / sum;
    #pragma unroll
    for (int e = 0; e < 8; ++e) chunk[e] = (bf16)(vals[e] * inv);
    *(v8bf*)(row + lane * 8) = chunk;
  }
  __syncthreads();

  head_mix(S, sT2, sB2);

  // ---- O_h(16x128) = attn_h(16x256) @ V_h^T : two V-fragment groups ping-pong ----
  const bf16* vh = V + ((size_t)b * DH + h * DHEAD) * NTOK;
  const bf16* srow = &S[(h * 16 + col) * NTOK];
  v8f acc[8] = {};
  v16bf a = load_afrag(srow, lane);
  v16bf bbA[4];
  #pragma unroll
  for (int d = 0; d < 4; ++d)
    bbA[d] = load_bfrag(vh + (size_t)(d * 16 + col) * NTOK, lane);
  #pragma unroll 1
  for (int kc = 0; kc < 8; ++kc) {
    const int m0 = kc * 32;
    v16bf bbB[4];
    #pragma unroll
    for (int d = 0; d < 4; ++d)
      bbB[d] = load_bfrag(vh + (size_t)((d + 4) * 16 + col) * NTOK + m0, lane);
    // group A WMMAs hide group B loads
    #pragma unroll
    for (int d = 0; d < 4; ++d)
      acc[d] = __builtin_amdgcn_wmma_f32_16x16x32_bf16(
          false, a, false, bbA[d], (short)0, acc[d], false, false);
    // refill group A for next chunk while group B computes
    if (kc < 7) {
      #pragma unroll
      for (int d = 0; d < 4; ++d)
        bbA[d] = load_bfrag(vh + (size_t)(d * 16 + col) * NTOK + m0 + 32, lane);
    }
    #pragma unroll
    for (int d = 0; d < 4; ++d)
      acc[d + 4] = __builtin_amdgcn_wmma_f32_16x16x32_bf16(
          false, a, false, bbB[d], (short)0, acc[d + 4], false, false);
    if (kc < 7) a = load_afrag(srow + m0 + 32, lane);  // LDS, cheap
  }
  // epilogue: +v_local, ReLU, token-major U
  #pragma unroll
  for (int dt = 0; dt < 8; ++dt) {
    #pragma unroll
    for (int r = 0; r < 8; ++r) {
      int nl = n0 + r + rhalf;
      int ch = h * DHEAD + dt * 16 + col;
      float o = acc[dt][r] + (float)vloc[((size_t)b * DH + ch) * NTOK + nl];
      U[((size_t)b * NTOK + nl) * DH + ch] = (bf16)fmaxf(o, 0.f);
    }
  }
}

// ------------------------------------------------------------------
extern "C" void kernel_launch(void* const* d_in, const int* in_sizes, int n_in,
                              void* d_out, int out_size, void* d_ws, size_t ws_size,
                              hipStream_t stream) {
  (void)n_in; (void)out_size; (void)ws_size;
  const float* x    = (const float*)d_in[0];
  const float* qw   = (const float*)d_in[1];
  const float* qb   = (const float*)d_in[2];
  const float* kw   = (const float*)d_in[3];
  const float* kb   = (const float*)d_in[4];
  const float* vw   = (const float*)d_in[5];
  const float* vb   = (const float*)d_in[6];
  const float* vlw  = (const float*)d_in[7];
  const float* vlb  = (const float*)d_in[8];
  const float* th1w = (const float*)d_in[9];
  const float* th1b = (const float*)d_in[10];
  const float* th2w = (const float*)d_in[11];
  const float* th2b = (const float*)d_in[12];
  const float* ab   = (const float*)d_in[13];
  const float* pw   = (const float*)d_in[14];
  const float* pb   = (const float*)d_in[15];
  const float* f1w  = (const float*)d_in[16];
  const float* f1b  = (const float*)d_in[17];
  const float* g1   = (const float*)d_in[18];
  const float* b1   = (const float*)d_in[19];
  const float* m1   = (const float*)d_in[20];
  const float* v1   = (const float*)d_in[21];
  const float* mw   = (const float*)d_in[22];
  const float* mb   = (const float*)d_in[23];
  const float* gm   = (const float*)d_in[24];
  const float* bm   = (const float*)d_in[25];
  const float* mm   = (const float*)d_in[26];
  const float* vm   = (const float*)d_in[27];
  const float* f2w  = (const float*)d_in[28];
  const float* f2b  = (const float*)d_in[29];
  const float* g2   = (const float*)d_in[30];
  const float* b2   = (const float*)d_in[31];
  const float* m2   = (const float*)d_in[32];
  const float* v2   = (const float*)d_in[33];
  const int*   bidx = (const int*)d_in[34];

  const int B = in_sizes[0] / (DIM * NTOK);

  char* wp = (char*)d_ws;
  auto alloc = [&](size_t bytes) -> void* {
    void* p = (void*)wp;
    wp += (bytes + 255) & ~(size_t)255;
    return p;
  };
  bf16* xt   = (bf16*)alloc((size_t)B * NTOK * DIM * 2);   // token-major x
  bf16* wqb  = (bf16*)alloc((size_t)256 * DIM * 2);
  bf16* wkb  = (bf16*)alloc((size_t)256 * DIM * 2);
  bf16* wvb  = (bf16*)alloc((size_t)DH * DIM * 2);
  bf16* wpwb = (bf16*)alloc((size_t)DIM * DH * 2);
  bf16* wf1b = (bf16*)alloc((size_t)HID * DIM * 2);
  bf16* wf2b = (bf16*)alloc((size_t)DIM * HID * 2);
  bf16* qt   = (bf16*)alloc((size_t)B * NTOK * 256 * 2);   // token-major Q
  bf16* kt   = (bf16*)alloc((size_t)B * NTOK * 256 * 2);   // token-major K
  bf16* vbuf = (bf16*)alloc((size_t)B * DH * NTOK * 2);    // channel-major V
  bf16* vloc = (bf16*)alloc((size_t)B * DH * NTOK * 2);    // channel-major v_local
  bf16* ubuf = (bf16*)alloc((size_t)B * NTOK * DH * 2);    // token-major U
  float* x1  = (float*)alloc((size_t)B * NTOK * DIM * 4);  // token-major f32
  bf16* x1t  = (bf16*)alloc((size_t)B * NTOK * DIM * 2);
  bf16* h1t  = (bf16*)alloc((size_t)B * NTOK * HID * 2);
  bf16* h2t  = (bf16*)alloc((size_t)B * NTOK * HID * 2);
  float* s1  = (float*)alloc(HID * 4);
  float* t1  = (float*)alloc(HID * 4);
  float* smd = (float*)alloc(HID * 4);
  float* tmd = (float*)alloc(HID * 4);
  float* s2  = (float*)alloc(DIM * 4);
  float* t2  = (float*)alloc(DIM * 4);

  auto cvt = [&](const float* src, bf16* dst, int n) {
    cvt_bf16<<<(n + 255) / 256, 256, 0, stream>>>(src, dst, n);
  };

  // Phase 0: conversions + BN folds
  {
    int total = B * NTOK * DIM;
    cvt_tr_bf16<<<(total + 255) / 256, 256, 0, stream>>>(x, xt, DIM, total);
  }
  cvt(qw, wqb, 256 * DIM);
  cvt(kw, wkb, 256 * DIM);
  cvt(vw, wvb, DH * DIM);
  cvt(pw, wpwb, DIM * DH);
  cvt(f1w, wf1b, HID * DIM);
  cvt(f2w, wf2b, DIM * HID);
  bn_fold<<<(HID + 255) / 256, 256, 0, stream>>>(f1b, g1, b1, m1, v1, s1, t1, HID);
  bn_fold<<<(HID + 255) / 256, 256, 0, stream>>>(mb, gm, bm, mm, vm, smd, tmd, HID);
  bn_fold<<<(DIM + 255) / 256, 256, 0, stream>>>(f2b, g2, b2, m2, v2, s2, t2, DIM);

  // Phase 1: Q, K (token-major), V (channel-major for attn@V / dwconv)
  gemm_bf16<<<dim3(4, 2, B), 256, 0, stream>>>(wqb, xt, qb, nullptr, nullptr,
                                               nullptr, qt, 256, DIM, 0, 0, 0, 0);
  gemm_bf16<<<dim3(4, 2, B), 256, 0, stream>>>(wkb, xt, kb, nullptr, nullptr,
                                               nullptr, kt, 256, DIM, 0, 0, 0, 0);
  gemm_bf16<<<dim3(4, DH / 128, B), 256, 0, stream>>>(wvb, xt, vb, nullptr, nullptr,
                                                      nullptr, vbuf, DH, DIM, 0, 1, 0, 0);

  // Phase 2: v_local = dw3x3(V) + bias (channel-major)
  {
    int total = B * DH * NTOK;
    dwconv3_c<<<(total + 255) / 256, 256, 0, stream>>>(vbuf, vlw, vlb, vloc, DH, total);
  }

  // Phase 3: fused attention -> U = relu(O + v_local), token-major bf16
  attn_fused<<<B * 16, 256, 0, stream>>>(qt, kt, vbuf, ab, bidx, th1w, th1b,
                                         th2w, th2b, vloc, ubuf);

  // Phase 4: x1 = x + (pw @ U + pb); x channel-major resid, outputs token-major
  gemm_bf16<<<dim3(4, DIM / 128, B), 256, 0, stream>>>(wpwb, ubuf, pb, nullptr, x,
                                                       x1, x1t, DIM, DH, 1, 0, 0, 1);

  // Phase 5: h1 = relu(bn(f1 @ x1)), token-major
  gemm_bf16<<<dim3(4, HID / 128, B), 256, 0, stream>>>(wf1b, x1t, s1, t1, nullptr,
                                                       nullptr, h1t, HID, DIM, 2, 0, 0, 0);

  // Phase 6: h2 = relu(bn(dw3x3(h1))), token-major
  {
    int total = B * NTOK * HID;
    dwconv3_t<<<(total + 255) / 256, 256, 0, stream>>>(h1t, mw, smd, tmd, h2t, HID, total);
  }

  // Phase 7: out = x1 + bn(f2 @ h2); final store channel-major f32 to d_out
  gemm_bf16<<<dim3(4, DIM / 128, B), 256, 0, stream>>>(wf2b, h2t, s2, t2, x1,
                                                       (float*)d_out, nullptr,
                                                       DIM, HID, 3, 0, 1, 0);
}
